// KumaDecompAttModel_43765716747116
// MI455X (gfx1250) — matrix operand order repaired and checked by reading, not compiled, not run on hardware
//
#include <hip/hip_runtime.h>
#include <hip/hip_bf16.h>

typedef __attribute__((ext_vector_type(16))) _Float16 v16h;
typedef __attribute__((ext_vector_type(8)))  _Float16 v8h;
typedef __attribute__((ext_vector_type(8)))  float    v8f;
typedef int v4i __attribute__((vector_size(16)));

#define NEGBIG (-1e9f)

#define AS1 __attribute__((address_space(1)))
#define AS3 __attribute__((address_space(3)))

#if defined(__gfx1250__) && __has_builtin(__builtin_amdgcn_global_load_async_to_lds_b128)
#define HAVE_ASYNC_LDS 1
#else
#define HAVE_ASYNC_LDS 0
#endif

__device__ __forceinline__ void wait_async0() {
#if HAVE_ASYNC_LDS
#if __has_builtin(__builtin_amdgcn_s_wait_asynccnt)
  __builtin_amdgcn_s_wait_asynccnt(0);
#else
  asm volatile("s_wait_asynccnt 0" ::: "memory");
#endif
#endif
}

#if HAVE_ASYNC_LDS
__device__ __forceinline__ void async_b128(const _Float16* g, _Float16* l) {
  __builtin_amdgcn_global_load_async_to_lds_b128(
      (AS1 v4i*)g, (AS3 v4i*)l, 0, 0);
}
#endif

// ---------------------------------------------------------------------------
// WMMA GEMM:  C[M,N] (f32 and/or f16) = act( A[M,K]f16 * B + bias )
//   TRANS_B = false : B is [K,N] row-major with leading dim ldb (weights)
//   TRANS_B = true  : B is [N,K] row-major with leading dim ldb (x @ y^T)
// Block tile 128(M) x 128(N) x 32(K); 256 threads = 8 waves (4M x 2N);
// each wave owns a 32x64 tile = 2x4 v_wmma_f32_16x16x32_f16 accumulators.
// A tile and transposed-B tile staged to LDS with GLOBAL_LOAD_ASYNC_TO_LDS
// (ASYNCcnt) when the builtin exists; sync global_load_b128 otherwise.
// Batched via blockIdx.z.  Requires: K % 32 == 0, lda/ldb % 8 == 0.
// ---------------------------------------------------------------------------
template<bool TRANS_B, bool RELU>
__global__ __launch_bounds__(256) void gemm_wmma(
    const _Float16* __restrict__ A, const _Float16* __restrict__ B,
    const float* __restrict__ bias,
    float* __restrict__ C, _Float16* __restrict__ C16,
    int M, int N, int K, int lda, int ldb, int ldc, int ldc16,
    long sA, long sB, long sC, long sC16)
{
  __shared__ __align__(16) _Float16 Al[128 * 40];  // 128 rows x 32 K (+8 pad)
  __shared__ __align__(16) _Float16 Bl[128 * 40];  // stored [n][k]

  const int tid  = threadIdx.x;
  const int bz   = blockIdx.z;
  A += (long)bz * sA;
  B += (long)bz * sB;
  if (C)   C   += (long)bz * sC;
  if (C16) C16 += (long)bz * sC16;

  const int m0   = blockIdx.y * 128;
  const int n0   = blockIdx.x * 128;
  const int wid  = tid >> 5;
  const int lane = tid & 31;
  const int wm   = (wid >> 1) * 32;   // 4 waves along M
  const int wn   = (wid & 1) * 64;    // 2 waves along N

  v8f acc[2][4] = {};

  for (int k0 = 0; k0 < K; k0 += 32) {
    __syncthreads();
#if HAVE_ASYNC_LDS
    // ---- A tile (128x32) via async DMA to LDS ----
    #pragma unroll
    for (int v = 0; v < 2; ++v) {
      int lin = tid + v * 256;
      int r = lin >> 2, c = (lin & 3) * 8;
      int gm = m0 + r;
      if (gm < M)
        async_b128(A + (long)gm * lda + k0 + c, Al + r * 40 + c);
    }
    if (TRANS_B) {
      #pragma unroll
      for (int v = 0; v < 2; ++v) {
        int lin = tid + v * 256;
        int r = lin >> 2, c = (lin & 3) * 8;
        int gn = n0 + r;
        if (gn < N)
          async_b128(B + (long)gn * ldb + k0 + c, Bl + r * 40 + c);
      }
    } else {
      // transpose staging cannot use per-lane async copies: sync load + ds_store
      #pragma unroll
      for (int v = 0; v < 2; ++v) {
        int lin = tid + v * 256;
        int k = lin >> 4, nv = (lin & 15) * 8;
        _Float16 tmp[8];
        if (n0 + nv + 8 <= N) {
          *(uint4*)tmp = *(const uint4*)(B + (long)(k0 + k) * ldb + n0 + nv);
        } else {
          #pragma unroll
          for (int i = 0; i < 8; ++i) {
            int gn = n0 + nv + i;
            tmp[i] = (gn < N) ? B[(long)(k0 + k) * ldb + gn] : (_Float16)0.f;
          }
        }
        #pragma unroll
        for (int i = 0; i < 8; ++i) Bl[(nv + i) * 40 + k] = tmp[i];
      }
    }
    wait_async0();
#else
    // ---- sync staging path ----
    #pragma unroll
    for (int v = 0; v < 2; ++v) {
      int lin = tid + v * 256;
      int r = lin >> 2, c = (lin & 3) * 8;
      uint4 d = make_uint4(0u, 0u, 0u, 0u);
      int gm = m0 + r;
      if (gm < M) d = *(const uint4*)(A + (long)gm * lda + k0 + c);
      *(uint4*)(Al + r * 40 + c) = d;
    }
    if (TRANS_B) {
      #pragma unroll
      for (int v = 0; v < 2; ++v) {
        int lin = tid + v * 256;
        int r = lin >> 2, c = (lin & 3) * 8;
        uint4 d = make_uint4(0u, 0u, 0u, 0u);
        int gn = n0 + r;
        if (gn < N) d = *(const uint4*)(B + (long)gn * ldb + k0 + c);
        *(uint4*)(Bl + r * 40 + c) = d;
      }
    } else {
      #pragma unroll
      for (int v = 0; v < 2; ++v) {
        int lin = tid + v * 256;
        int k = lin >> 4, nv = (lin & 15) * 8;
        _Float16 tmp[8];
        if (n0 + nv + 8 <= N) {
          *(uint4*)tmp = *(const uint4*)(B + (long)(k0 + k) * ldb + n0 + nv);
        } else {
          #pragma unroll
          for (int i = 0; i < 8; ++i) {
            int gn = n0 + nv + i;
            tmp[i] = (gn < N) ? B[(long)(k0 + k) * ldb + gn] : (_Float16)0.f;
          }
        }
        #pragma unroll
        for (int i = 0; i < 8; ++i) Bl[(nv + i) * 40 + k] = tmp[i];
      }
    }
    if (k0 + 32 < K) {  // prefetch next A tile (global_prefetch_b8)
      int gm = m0 + (tid >> 2);
      if (gm < M) __builtin_prefetch(A + (long)gm * lda + k0 + 32, 0, 0);
    }
#endif
    __syncthreads();

    // ---- build fragments per ISA VGPR layouts ----
    v16h af[2], bf[4];
    #pragma unroll
    for (int mi = 0; mi < 2; ++mi) {
      int row = wm + mi * 16 + (lane & 15);
      int kb  = (lane >> 4) * 8;                 // lanes 0-15: K0-7/16-23; 16-31: K8-15/24-31
      v8h lo = *(const v8h*)(Al + row * 40 + kb);
      v8h hi = *(const v8h*)(Al + row * 40 + kb + 16);
      #pragma unroll
      for (int i = 0; i < 8; ++i) { af[mi][i] = lo[i]; af[mi][i + 8] = hi[i]; }
    }
    #pragma unroll
    for (int ni = 0; ni < 4; ++ni) {
      int col = wn + ni * 16 + (lane & 15);
      int kb  = (lane >> 4) * 8;                 // lanes 0-15: K0-7/16-23; 16-31: K8-15/24-31
      v8h lo = *(const v8h*)(Bl + col * 40 + kb);
      v8h hi = *(const v8h*)(Bl + col * 40 + kb + 16);
      #pragma unroll
      for (int i = 0; i < 8; ++i) { bf[ni][i] = lo[i]; bf[ni][i + 8] = hi[i]; }
    }
    #pragma unroll
    for (int mi = 0; mi < 2; ++mi)
      #pragma unroll
      for (int ni = 0; ni < 4; ++ni)
        acc[mi][ni] = __builtin_amdgcn_wmma_f32_16x16x32_f16(
            false, af[mi], false, bf[ni], (short)0, acc[mi][ni], false, false);
  }

  // ---- epilogue: C/D layout: lanes 0-15 -> M=v, N=lane; 16-31 -> M=8+v ----
  #pragma unroll
  for (int mi = 0; mi < 2; ++mi)
    #pragma unroll
    for (int ni = 0; ni < 4; ++ni) {
      int col   = n0 + wn + ni * 16 + (lane & 15);
      int rbase = m0 + wm + mi * 16 + ((lane >> 4) * 8);
      if (col < N) {
        float bv = bias ? bias[col] : 0.f;
        #pragma unroll
        for (int v = 0; v < 8; ++v) {
          int row = rbase + v;
          if (row < M) {
            float x = acc[mi][ni][v] + bv;
            if (RELU) x = x > 0.f ? x : 0.f;
            if (C)   C[(long)row * ldc + col] = x;
            if (C16) C16[(long)row * ldc16 + col] = (_Float16)x;
          }
        }
      }
    }
}

// ---------------------------------------------------------------------------
// Row softmax over 256 keys. SELF adds relative-distance bias + diagonal mask.
// TRANSP reads S transposed (for h2p). Writes f16 probabilities.
// ---------------------------------------------------------------------------
template<bool SELF, bool TRANSP>
__global__ __launch_bounds__(256) void softmax_rows(
    const float* __restrict__ S, const float* __restrict__ distW,
    const int* __restrict__ keymask, _Float16* __restrict__ P, int L)
{
  __shared__ float red[256];
  int q = blockIdx.x, b = blockIdx.y, k = threadIdx.x;
  long base = (long)b * L * L;
  float v = TRANSP ? S[base + (long)k * L + q] : S[base + (long)q * L + k];
  bool mk = keymask[b * L + k] != 0;
  if (SELF) {
    int d = k - q; d = d < -11 ? -11 : (d > 11 ? 11 : d);
    v += distW[d + 11];
    if (k == q) v = NEGBIG;
  }
  if (!mk) v = NEGBIG;
  red[k] = v; __syncthreads();
  for (int s = 128; s > 0; s >>= 1) { if (k < s) red[k] = fmaxf(red[k], red[k + s]); __syncthreads(); }
  float mx = red[0]; __syncthreads();
  float e = __expf(v - mx);
  red[k] = e; __syncthreads();
  for (int s = 128; s > 0; s >>= 1) { if (k < s) red[k] += red[k + s]; __syncthreads(); }
  float sum = red[0];
  P[base + (long)q * L + k] = (_Float16)(e / sum);
}

// ---------------------------------------------------------------------------
__global__ void gather_embed(const int* __restrict__ idx, const float* __restrict__ embW,
                             _Float16* __restrict__ out, int* __restrict__ mask, int E)
{
  int t = blockIdx.x;
  int id = idx[t];
  int m = (id != 1);
  if (threadIdx.x == 0) mask[t] = m;
  const float* src = embW + (long)id * E;
  for (int e = threadIdx.x; e < E; e += blockDim.x)
    out[(long)t * E + e] = m ? (_Float16)src[e] : (_Float16)0.f;
}

__global__ void f32_to_f16(const float* __restrict__ in, _Float16* __restrict__ out, long n)
{
  long i = (long)blockIdx.x * blockDim.x + threadIdx.x;
  if (i < n) out[i] = (_Float16)in[i];
}

__global__ void copy_rows_f16(const _Float16* __restrict__ src, _Float16* __restrict__ dst,
                              int cols, int ldsrc, int lddst)
{
  long r = blockIdx.x;
  for (int c = threadIdx.x; c < cols; c += blockDim.x)
    dst[r * lddst + c] = src[r * ldsrc + c];
}

__global__ void masked_sum(const float* __restrict__ src, const int* __restrict__ mask,
                           _Float16* __restrict__ dst, int L, int Hh, int lddst, int coff)
{
  int b = blockIdx.x;
  int h = blockIdx.y * blockDim.x + threadIdx.x;
  if (h >= Hh) return;
  float a = 0.f;
  for (int l = 0; l < L; ++l)
    if (mask[b * L + l]) a += src[((long)b * L + l) * Hh + h];
  dst[(long)b * lddst + coff + h] = (_Float16)a;
}

__global__ void final_out(const _Float16* __restrict__ agg, const float* __restrict__ W,
                          float* __restrict__ out, int Hh, int O)
{
  int b = blockIdx.x, lane = threadIdx.x;  // 32 threads (one wave)
  for (int o = 0; o < O; ++o) {
    float a = 0.f;
    for (int k = lane; k < Hh; k += 32) a += (float)agg[(long)b * Hh + k] * W[(long)k * O + o];
    for (int s = 16; s > 0; s >>= 1) a += __shfl_xor(a, s, 32);
    if (lane == 0) out[b * O + o] = a;
  }
}

// ---------------------------------------------------------------------------
static void gemm(hipStream_t st, bool transB, bool relu,
                 const _Float16* A, const _Float16* Bm, const float* bias,
                 float* C, _Float16* C16, int M, int N, int K,
                 int lda, int ldb, int ldc, int ldc16,
                 long sA, long sB, long sC, long sC16, int batches)
{
  dim3 g((unsigned)((N + 127) / 128), (unsigned)((M + 127) / 128), (unsigned)batches);
  dim3 blk(256);
  if (transB) {
    if (relu) hipLaunchKernelGGL((gemm_wmma<true,  true >), g, blk, 0, st, A, Bm, bias, C, C16, M, N, K, lda, ldb, ldc, ldc16, sA, sB, sC, sC16);
    else      hipLaunchKernelGGL((gemm_wmma<true,  false>), g, blk, 0, st, A, Bm, bias, C, C16, M, N, K, lda, ldb, ldc, ldc16, sA, sB, sC, sC16);
  } else {
    if (relu) hipLaunchKernelGGL((gemm_wmma<false, true >), g, blk, 0, st, A, Bm, bias, C, C16, M, N, K, lda, ldb, ldc, ldc16, sA, sB, sC, sC16);
    else      hipLaunchKernelGGL((gemm_wmma<false, false>), g, blk, 0, st, A, Bm, bias, C, C16, M, N, K, lda, ldb, ldc, ldc16, sA, sB, sC, sC16);
  }
}

extern "C" void kernel_launch(void* const* d_in, const int* in_sizes, int n_in,
                              void* d_out, int out_size, void* d_ws, size_t ws_size,
                              hipStream_t stream)
{
  (void)in_sizes; (void)n_in; (void)out_size; (void)ws_size;
  const int*   premI = (const int*)d_in[0];
  const int*   hypoI = (const int*)d_in[1];
  const float* embW  = (const float*)d_in[2];
  const float* projW = (const float*)d_in[3];
  const float* projB = (const float*)d_in[4];
  const float* distW = (const float*)d_in[5];
  const float* saW1  = (const float*)d_in[6],  *saB1  = (const float*)d_in[7];
  const float* saW2  = (const float*)d_in[8],  *saB2  = (const float*)d_in[9];
  const float* attW1 = (const float*)d_in[10], *attB1 = (const float*)d_in[11];
  const float* attW2 = (const float*)d_in[12], *attB2 = (const float*)d_in[13];
  const float* cmpW1 = (const float*)d_in[14], *cmpB1 = (const float*)d_in[15];
  const float* cmpW2 = (const float*)d_in[16], *cmpB2 = (const float*)d_in[17];
  const float* aggW1 = (const float*)d_in[18], *aggB1 = (const float*)d_in[19];
  const float* aggW2 = (const float*)d_in[20], *aggB2 = (const float*)d_in[21];
  const float* outW  = (const float*)d_in[22];

  constexpr int Bn = 64, L = 256, E = 1024, P = 512, H = 512, D2 = 1024, O = 3;
  constexpr int NT = Bn * L;                    // 16384 tokens per side

  char* ws = (char*)d_ws;
  size_t off = 0;
  auto alloc = [&](size_t bytes) -> char* {
    char* p = ws + off; off += (bytes + 255) & ~size_t(255); return p;
  };

  // f16 weight copies ([K,N] row-major)
  _Float16* w_proj = (_Float16*)alloc((size_t)E  * P * 2);
  _Float16* w_sa1  = (_Float16*)alloc((size_t)P  * H * 2);
  _Float16* w_sa2  = (_Float16*)alloc((size_t)H  * H * 2);
  _Float16* w_att1 = (_Float16*)alloc((size_t)D2 * H * 2);
  _Float16* w_att2 = (_Float16*)alloc((size_t)H  * H * 2);
  _Float16* w_cmp1 = (_Float16*)alloc((size_t)2 * D2 * H * 2);
  _Float16* w_cmp2 = (_Float16*)alloc((size_t)H  * H * 2);
  _Float16* w_agg1 = (_Float16*)alloc((size_t)2 * H * H * 2);
  _Float16* w_agg2 = (_Float16*)alloc((size_t)H  * H * 2);
  // activations
  _Float16* emb16  = (_Float16*)alloc((size_t)NT * E * 2);      // reused prem->hypo
  int*      maskP  = (int*)alloc((size_t)NT * 4);
  int*      maskH  = (int*)alloc((size_t)NT * 4);
  _Float16* x16p   = (_Float16*)alloc((size_t)NT * P * 2);
  _Float16* x16h   = (_Float16*)alloc((size_t)NT * P * 2);
  _Float16* h1     = (_Float16*)alloc((size_t)NT * H * 2);      // shared hidden scratch
  _Float16* p16    = (_Float16*)alloc((size_t)NT * H * 2);      // self-att MLP out (per side)
  float*    sf32   = (float*)alloc((size_t)Bn * L * L * 4);     // scores (per side)
  _Float16* prob16 = (_Float16*)alloc((size_t)Bn * L * L * 2);  // self-att probs (per side)
  _Float16* x2p    = (_Float16*)alloc((size_t)NT * D2 * 2);     // [x, a@x] prem
  _Float16* x2h    = (_Float16*)alloc((size_t)NT * D2 * 2);     // [x, a@x] hypo
  _Float16* pq16   = (_Float16*)alloc((size_t)NT * H * 2);
  _Float16* hk16   = (_Float16*)alloc((size_t)NT * H * 2);
  float*    simf   = (float*)alloc((size_t)Bn * L * L * 4);
  _Float16* p2h16  = (_Float16*)alloc((size_t)Bn * L * L * 2);
  _Float16* h2p16  = (_Float16*)alloc((size_t)Bn * L * L * 2);
  _Float16* cmpinP = (_Float16*)alloc((size_t)NT * 2 * D2 * 2); // [prem2, att_hypo]
  _Float16* cmpinH = (_Float16*)alloc((size_t)NT * 2 * D2 * 2); // [hypo2, att_prem]
  float*    cmpout = (float*)alloc((size_t)NT * H * 4);         // reused prem->hypo
  _Float16* aggin  = (_Float16*)alloc((size_t)Bn * 2 * H * 2);
  _Float16* aggh1  = (_Float16*)alloc((size_t)Bn * H * 2);
  _Float16* agg2   = (_Float16*)alloc((size_t)Bn * H * 2);

  auto conv = [&](const float* s, _Float16* d, long n) {
    f32_to_f16<<<(unsigned)((n + 255) / 256), 256, 0, stream>>>(s, d, n);
  };
  conv(projW, w_proj, (long)E * P);
  conv(saW1,  w_sa1,  (long)P * H);   conv(saW2,  w_sa2,  (long)H * H);
  conv(attW1, w_att1, (long)D2 * H);  conv(attW2, w_att2, (long)H * H);
  conv(cmpW1, w_cmp1, (long)2 * D2 * H); conv(cmpW2, w_cmp2, (long)H * H);
  conv(aggW1, w_agg1, (long)2 * H * H);  conv(aggW2, w_agg2, (long)H * H);

  // ---- embedding + projection ----
  gather_embed<<<NT, 256, 0, stream>>>(premI, embW, emb16, maskP, E);
  gemm(stream, false, false, emb16, w_proj, projB, nullptr, x16p,
       NT, P, E, E, P, 0, P, 0, 0, 0, 0, 1);
  gather_embed<<<NT, 256, 0, stream>>>(hypoI, embW, emb16, maskH, E);
  gemm(stream, false, false, emb16, w_proj, projB, nullptr, x16h,
       NT, P, E, E, P, 0, P, 0, 0, 0, 0, 1);

  // ---- self attention (per side) ----
  struct Side { const _Float16* x; const int* mask; _Float16* x2; };
  Side sides[2] = { { x16p, maskP, x2p }, { x16h, maskH, x2h } };
  for (int s = 0; s < 2; ++s) {
    gemm(stream, false, true, sides[s].x, w_sa1, saB1, nullptr, h1,
         NT, H, P, P, H, 0, H, 0, 0, 0, 0, 1);
    gemm(stream, false, true, h1, w_sa2, saB2, nullptr, p16,
         NT, H, H, H, H, 0, H, 0, 0, 0, 0, 1);
    // scores = p @ p^T  (batched)
    gemm(stream, true, false, p16, p16, nullptr, sf32, nullptr,
         L, L, H, H, H, L, 0, (long)L * H, (long)L * H, (long)L * L, 0, Bn);
    softmax_rows<true, false><<<dim3(L, Bn), L, 0, stream>>>(sf32, distW, sides[s].mask, prob16, L);
    // a @ x  -> right half of x2
    gemm(stream, false, false, prob16, sides[s].x, nullptr, nullptr, sides[s].x2 + P,
         L, P, L, L, P, 0, D2, (long)L * L, (long)L * P, 0, (long)L * D2, Bn);
    copy_rows_f16<<<NT, 256, 0, stream>>>(sides[s].x, sides[s].x2, P, P, D2);
  }

  // ---- attend MLP ----
  gemm(stream, false, true, x2p, w_att1, attB1, nullptr, h1, NT, H, D2, D2, H, 0, H, 0, 0, 0, 0, 1);
  gemm(stream, false, true, h1, w_att2, attB2, nullptr, pq16, NT, H, H, H, H, 0, H, 0, 0, 0, 0, 1);
  gemm(stream, false, true, x2h, w_att1, attB1, nullptr, h1, NT, H, D2, D2, H, 0, H, 0, 0, 0, 0, 1);
  gemm(stream, false, true, h1, w_att2, attB2, nullptr, hk16, NT, H, H, H, H, 0, H, 0, 0, 0, 0, 1);

  // ---- cross attention ----
  gemm(stream, true, false, pq16, hk16, nullptr, simf, nullptr,
       L, L, H, H, H, L, 0, (long)L * H, (long)L * H, (long)L * L, 0, Bn);
  softmax_rows<false, false><<<dim3(L, Bn), L, 0, stream>>>(simf, nullptr, maskH, p2h16, L);
  softmax_rows<false, true ><<<dim3(L, Bn), L, 0, stream>>>(simf, nullptr, maskP, h2p16, L);
  // attended_hypo = p2h @ hypo2 -> right half of cmpinP ; left half = prem2
  gemm(stream, false, false, p2h16, x2h, nullptr, nullptr, cmpinP + D2,
       L, D2, L, L, D2, 0, 2 * D2, (long)L * L, (long)L * D2, 0, (long)L * 2 * D2, Bn);
  copy_rows_f16<<<NT, 256, 0, stream>>>(x2p, cmpinP, D2, D2, 2 * D2);
  // attended_prem = h2p @ prem2 -> right half of cmpinH ; left half = hypo2
  gemm(stream, false, false, h2p16, x2p, nullptr, nullptr, cmpinH + D2,
       L, D2, L, L, D2, 0, 2 * D2, (long)L * L, (long)L * D2, 0, (long)L * 2 * D2, Bn);
  copy_rows_f16<<<NT, 256, 0, stream>>>(x2h, cmpinH, D2, D2, 2 * D2);

  // ---- compare MLP + masked length-sum ----
  gemm(stream, false, true, cmpinP, w_cmp1, cmpB1, nullptr, h1, NT, H, 2 * D2, 2 * D2, H, 0, H, 0, 0, 0, 0, 1);
  gemm(stream, false, true, h1, w_cmp2, cmpB2, cmpout, nullptr, NT, H, H, H, H, H, 0, 0, 0, 0, 0, 1);
  masked_sum<<<dim3(Bn, 2), 256, 0, stream>>>(cmpout, maskP, aggin, L, H, 2 * H, 0);
  gemm(stream, false, true, cmpinH, w_cmp1, cmpB1, nullptr, h1, NT, H, 2 * D2, 2 * D2, H, 0, H, 0, 0, 0, 0, 1);
  gemm(stream, false, true, h1, w_cmp2, cmpB2, cmpout, nullptr, NT, H, H, H, H, H, 0, 0, 0, 0, 0, 1);
  masked_sum<<<dim3(Bn, 2), 256, 0, stream>>>(cmpout, maskH, aggin, L, H, 2 * H, H);

  // ---- aggregate MLP + output ----
  gemm(stream, false, true, aggin, w_agg1, aggB1, nullptr, aggh1, Bn, H, 2 * H, 2 * H, H, 0, H, 0, 0, 0, 0, 1);
  gemm(stream, false, true, aggh1, w_agg2, aggB2, nullptr, agg2, Bn, H, H, H, H, 0, H, 0, 0, 0, 0, 1);
  final_out<<<Bn, 32, 0, stream>>>(agg2, outW, (float*)d_out, H, O);
}
